// PK_RNN_11287174053959
// MI455X (gfx1250) — matrix operand order repaired and checked by reading, not compiled, not run on hardware
//
#include <hip/hip_runtime.h>
#include <hip/hip_bf16.h>
#include <stdint.h>

#define VOCAB   829
#define EMBED   8
#define CONTD   40
#define HID     32
#define BSZ     2048
#define TSZ     256
#define CODES   32
#define IN_DIM  49     // EMBED + CONT + 1
#define KPAD    64     // IN_DIM padded to 2x K=32 WMMA chunks
#define SCALEF  1000.0f

typedef __bf16 v16bf __attribute__((ext_vector_type(16)));
typedef float  v8f   __attribute__((ext_vector_type(8)));
typedef unsigned int v4u __attribute__((ext_vector_type(4)));
typedef int          v8i __attribute__((ext_vector_type(8)));
typedef int          v4i __attribute__((ext_vector_type(4)));

#if __has_builtin(__builtin_amdgcn_tensor_load_to_lds) && \
    __has_builtin(__builtin_amdgcn_s_wait_tensorcnt)
#define USE_TDM 1
#else
#define USE_TDM 0
#endif

__device__ __forceinline__ v8f wmma_bf16(v16bf a, v16bf b, v8f c) {
  // D = A(16x32 bf16) * B(32x16 bf16) + C(16x16 f32)
  return __builtin_amdgcn_wmma_f32_16x16x32_bf16(
      false, a, false, b, (short)0, c, false, false);
}

__device__ __forceinline__ float fast_rcp(float x) {
#if __has_builtin(__builtin_amdgcn_rcpf)
  return __builtin_amdgcn_rcpf(x);   // v_rcp_f32
#else
  return 1.0f / x;
#endif
}

__device__ __forceinline__ float sigmoidf_fast(float x) {
  return fast_rcp(1.0f + __expf(-x));
}

__device__ __forceinline__ float tanhf_fast(float x) {
#if __has_builtin(__builtin_amdgcn_tanhf)
  return __builtin_amdgcn_tanhf(x);  // v_tanh_f32 (gfx1250 trans op)
#else
  const float e = __expf(2.0f * x);
  return (e - 1.0f) * fast_rcp(e + 1.0f);
#endif
}

#if USE_TDM
// Tensor Data Mover: load one 16-row x 40-float cont tile into LDS with
// iterate mode so each batch row lands at an LDS stride of KPAD floats.
// 6-arg builtin: (group0, group1, group2, group3, group4(zero), cpol)
__device__ __forceinline__ void tdm_load_cont(const float* gsrc, unsigned lds_off) {
  const unsigned long long ga = (unsigned long long)(size_t)gsrc;
  v4u g0;
  g0[0] = 1u;                                   // count=1 (valid user D#)
  g0[1] = lds_off;                              // lds_addr (bytes)
  g0[2] = (unsigned)(ga & 0xFFFFFFFFu);         // global_addr[31:0]
  g0[3] = (unsigned)((ga >> 32) & 0x1FFFFFFu)   // global_addr[56:32]
        | (2u << 30);                           // type=2 ("image")
  v8i g1;
  g1[0] = (2 << 16) | (1 << 19);                // data_size=4B, iterate_enable=1
  g1[1] = (CONTD << 16);                        // tensor_dim0 = 40 (low16)
  g1[2] = (1 << 16);                            // tensor_dim0 hi=0 | tensor_dim1=1
  g1[3] = (CONTD << 16);                        // tensor_dim1 hi=0 | tile_dim0=40
  g1[4] = 1;                                    // tile_dim1=1, tile_dim2=0
  g1[5] = TSZ * CONTD;                          // tensor_dim0_stride low32
  g1[6] = 0;
  g1[7] = 0;
  v4i g2;
  g2[0] = 0;
  g2[1] = KPAD;                                 // lds_addr_increment (elements)
  g2[2] = TSZ * CONTD;                          // global_addr_increment low32
  g2[3] = (15 << 16);                           // iterate_count=15 -> 16 rows
  v4i g3 = {0, 0, 0, 0};
  v8i g4 = {0, 0, 0, 0, 0, 0, 0, 0};
  __builtin_amdgcn_tensor_load_to_lds(g0, g1, g2, g3, g4, 0);
}
#endif

__global__ void __launch_bounds__(32)
pk_gru_scan(const float* __restrict__ cont,       // (B,T,CONT)
            const int*   __restrict__ cat,        // (B,T,CODES)
            const float* __restrict__ lb,         // (B,T)
            const float* __restrict__ dose,       // (B,T)
            const float* __restrict__ td,         // (B,T)
            const float* __restrict__ vfw,        // (B,T) v_from_weight
            const float* __restrict__ vel,        // (B,T) vanco_el
            const float* __restrict__ embed,      // (VOCAB,EMBED)
            const float* __restrict__ W_ih,       // (96,49)
            const float* __restrict__ W_hh,       // (96,32)
            const float* __restrict__ b_ih,       // (96)
            const float* __restrict__ b_hh,       // (96)
            const float* __restrict__ W_out,      // (2,32)
            const float* __restrict__ b_out,      // (2)
            float* __restrict__ outK,             // (B,T)
            float* __restrict__ outV,             // (B,T)
            float* __restrict__ outC,             // (B,T)
            float* __restrict__ ws)               // 8 accumulators
{
  __shared__ float xbuf[2][16][KPAD];  // double-buffered input rows (8KB)
  __shared__ float hbuf[16][HID];      // hidden state staging for transpose
  __shared__ float kvraw[2][16];       // raw k/v dots per row

  const int lane  = threadIdx.x;       // one wave32 per block
  const int b0    = blockIdx.x * 16;   // 16 batch rows per wave
  const int ncol  = lane & 15;         // N-column / A-row this lane owns
  const int hiK   = (lane >> 4) * 16;  // B-operand: high lanes carry K+16
  const int loA   = (lane >> 4) * 8;   // unused: A-operand also packs with hiK (K+16)
  const int mbase = (lane >> 4) * 8;   // D-layout: high lanes carry rows 8..15

  // ---- pack weights into registers as WMMA B-operands (resident all kernel)
  v16bf Wih[2][6];
  v16bf Whh[6];
  v16bf WoutB;
#pragma unroll
  for (int t = 0; t < 6; ++t) {
    const int n = 16 * t + ncol;
#pragma unroll
    for (int c = 0; c < 2; ++c) {
#pragma unroll
      for (int e = 0; e < 16; ++e) {
        const int k = 32 * c + hiK + e;
        Wih[c][t][e] = (__bf16)((k < IN_DIM) ? W_ih[n * IN_DIM + k] : 0.0f);
      }
    }
#pragma unroll
    for (int e = 0; e < 16; ++e)
      Whh[t][e] = (__bf16)W_hh[n * HID + hiK + e];
  }
#pragma unroll
  for (int e = 0; e < 16; ++e)
    WoutB[e] = (__bf16)((ncol < 2) ? W_out[ncol * HID + hiK + e] : 0.0f);

  // ---- per-lane bias splats (column n = 16*t + ncol, same for both halves)
  float brz[4], bin[2], bhn[2];
#pragma unroll
  for (int t = 0; t < 4; ++t) brz[t] = b_ih[16 * t + ncol] + b_hh[16 * t + ncol];
#pragma unroll
  for (int t = 0; t < 2; ++t) {
    bin[t] = b_ih[64 + 16 * t + ncol];
    bhn[t] = b_hh[64 + 16 * t + ncol];
  }
  const float bkv = (ncol < 2) ? b_out[ncol] : 0.0f;

  // ---- LDS init: zero vfb/pad columns of both buffers, zero hidden state
  for (int i = lane; i < 16 * HID; i += 32) ((float*)hbuf)[i] = 0.0f;
  for (int s = 0; s < 2; ++s)
    for (int i = lane; i < 16 * (KPAD - IN_DIM + 1); i += 32) {
      const int r = i / (KPAD - IN_DIM + 1);
      const int c = i % (KPAD - IN_DIM + 1);
      xbuf[s][r][IN_DIM - 1 + c] = 0.0f;
    }
  __syncthreads();

#if USE_TDM
  const unsigned xlds[2] = {
      (unsigned)(size_t)&xbuf[0][0][EMBED],
      (unsigned)(size_t)&xbuf[1][0][EMBED]};
  tdm_load_cont(cont + (long)b0 * TSZ * CONTD, xlds[0]);   // tile for t=0
#endif

  // ---- recurrent state
  v16bf hA;                    // h(t-1) in A-layout (bf16), carried across steps
#pragma unroll
  for (int e = 0; e < 16; ++e) hA[e] = (__bf16)0.0f;
  float hD[2][8];              // h(t-1) in D-layout (f32)
#pragma unroll
  for (int tt = 0; tt < 2; ++tt)
#pragma unroll
    for (int u = 0; u < 8; ++u) hD[tt][u] = 0.0f;

  float tm = 0.0f, concn = 0.0f, prev_lb = 0.0f;
  float prev_k = 0.0f, prev_v = 0.0f;
  float a_mse = 0.0f, a_msk = 0.0f, a_dk = 0.0f, a_dv = 0.0f;
  float a_k0 = 0.0f, a_v0 = 0.0f;

  for (int t = 0; t < TSZ; ++t) {
    const int buf = t & 1;

#if USE_TDM
    if (t + 1 < TSZ)   // kick off next timestep's cont tile (async DMA)
      tdm_load_cont(cont + ((long)b0 * TSZ + (t + 1)) * CONTD, xlds[1 - buf]);
#else
    for (int i = lane; i < 16 * CONTD; i += 32) {
      const int r = i / CONTD, c = i % CONTD;
      xbuf[buf][r][EMBED + c] = cont[((long)(b0 + r) * TSZ + t) * CONTD + c];
    }
#endif

    // vfb feedback: previous concn + t-1 label
    if (lane < 16) {
      const float fm = (prev_lb != 0.0f) ? 1.0f : 0.0f;
      xbuf[buf][lane][IN_DIM - 1] = concn * (1.0f - fm) + prev_lb * fm;
    }

    // embedding gather: lane handles (row = lane&15, 16 of 32 codes)
    {
      const int* cp = cat + ((long)(b0 + ncol) * TSZ + t) * CODES + (lane >> 4) * 16;
      float4 p0 = {0, 0, 0, 0}, p1 = {0, 0, 0, 0};
#pragma unroll
      for (int c = 0; c < 16; ++c) {
        const float4* er = (const float4*)(embed + cp[c] * EMBED);
        const float4 e0 = er[0], e1 = er[1];
        p0.x += e0.x; p0.y += e0.y; p0.z += e0.z; p0.w += e0.w;
        p1.x += e1.x; p1.y += e1.y; p1.z += e1.z; p1.w += e1.w;
      }
      // combine the two 16-code halves across lane pairs (L, L^16)
      p0.x += __shfl_xor(p0.x, 16, 32); p0.y += __shfl_xor(p0.y, 16, 32);
      p0.z += __shfl_xor(p0.z, 16, 32); p0.w += __shfl_xor(p0.w, 16, 32);
      p1.x += __shfl_xor(p1.x, 16, 32); p1.y += __shfl_xor(p1.y, 16, 32);
      p1.z += __shfl_xor(p1.z, 16, 32); p1.w += __shfl_xor(p1.w, 16, 32);
      if (lane < 16) {
        *(float4*)&xbuf[buf][lane][0] = p0;
        *(float4*)&xbuf[buf][lane][4] = p1;
      }
    }

    if (t + 1 < TSZ)
      __builtin_prefetch(&cat[((long)(b0 + ncol) * TSZ + t + 1) * CODES], 0, 3);

#if USE_TDM
    if (t + 1 < TSZ) __builtin_amdgcn_s_wait_tensorcnt(1);  // tile t done
    else             __builtin_amdgcn_s_wait_tensorcnt(0);
#endif
    __syncthreads();   // [A] xbuf(t) fully populated

    // ---- pack x A-operands (f32 LDS -> bf16, contiguous K order k = 16h + e, matching B)
    v16bf xA0, xA1;
    {
      const float* xr = xbuf[buf][ncol];
#pragma unroll
      for (int e = 0; e < 16; ++e) {
        xA0[e] = (__bf16)xr[hiK + e];
        xA1[e] = (__bf16)xr[32 + hiK + e];
      }
    }

    // ---- 18 WMMAs: r/z gates accumulate gi+gh; n-gate keeps gh separate
    v8f acc[6], achn[2];
#pragma unroll
    for (int tt = 0; tt < 4; ++tt) {
      v8f a;
#pragma unroll
      for (int e = 0; e < 8; ++e) a[e] = brz[tt];
      a = wmma_bf16(xA0, Wih[0][tt], a);
      a = wmma_bf16(xA1, Wih[1][tt], a);
      a = wmma_bf16(hA,  Whh[tt],    a);
      acc[tt] = a;
    }
#pragma unroll
    for (int tt = 0; tt < 2; ++tt) {
      v8f a;
#pragma unroll
      for (int e = 0; e < 8; ++e) a[e] = bin[tt];
      a = wmma_bf16(xA0, Wih[0][4 + tt], a);
      a = wmma_bf16(xA1, Wih[1][4 + tt], a);
      acc[4 + tt] = a;
      v8f hh;
#pragma unroll
      for (int e = 0; e < 8; ++e) hh[e] = bhn[tt];
      achn[tt] = wmma_bf16(hA, Whh[4 + tt], hh);
    }

    // ---- gate nonlinearity + hidden update (D layout: elem=row, lane=col)
#pragma unroll
    for (int tt = 0; tt < 2; ++tt) {
#pragma unroll
      for (int u = 0; u < 8; ++u) {
        const float r  = sigmoidf_fast(acc[tt][u]);
        const float z  = sigmoidf_fast(acc[2 + tt][u]);
        const float nn = tanhf_fast(acc[4 + tt][u] + r * achn[tt][u]);
        hD[tt][u] = (1.0f - z) * nn + z * hD[tt][u];
      }
    }
#pragma unroll
    for (int tt = 0; tt < 2; ++tt)
#pragma unroll
      for (int u = 0; u < 8; ++u)
        hbuf[mbase + u][16 * tt + ncol] = hD[tt][u];
    __syncthreads();   // [B] h(t) staged for transpose

    // ---- repack h(t) as A-operand (reused next step for gh) + kv WMMA head
    {
      const float* hr = hbuf[ncol];
#pragma unroll
      for (int e = 0; e < 16; ++e) {
        hA[e] = (__bf16)hr[hiK + e];
      }
    }
    {
      v8f c;
#pragma unroll
      for (int e = 0; e < 8; ++e) c[e] = bkv;
      const v8f kv = wmma_bf16(hA, WoutB, c);   // cols 0/1 = k/v dots
      if (ncol < 2) {
#pragma unroll
        for (int u = 0; u < 8; ++u) kvraw[ncol][mbase + u] = kv[u];
      }
    }
    __syncthreads();   // [C] kvraw ready

    // ---- PK recursion + outputs + loss partials (lanes 0..15, row = lane)
    if (lane < 16) {
      const float k = __expf(kvraw[0][lane]);
      const float v = __expf(kvraw[1][lane]);
      const long bt = (long)(b0 + lane) * TSZ + t;
      const float tdv = td[bt], dd = dose[bt], lbv = lb[bt];
      const float Ae = __expf(-k * tdv);
      const float Bq = fast_rcp(k) * (1.0f - __expf(-k * dd)) * __expf(-k * (tdv - dd));
      tm = tm * Ae + Bq;
      concn = tm * fast_rcp(v + 1e-6f);
      outK[bt] = k; outV[bt] = v; outC[bt] = concn;

      const float mask = (lbv != 0.0f) ? 1.0f : 0.0f;
      const float d = (concn - lbv) * mask;
      a_mse += d * d;
      a_msk += mask;
      if (t == 0) {
        const float k0 = vel[bt], v0 = vfw[bt];
        a_k0 = (k0 - k) * (k0 - k);
        a_v0 = (v0 - v) * (v0 - v);
      } else {
        const float dk = k - prev_k, dv = v - prev_v;
        a_dk += dk * dk;
        a_dv += dv * dv;
      }
      prev_k = k; prev_v = v; prev_lb = lbv;
    }
  }

  if (lane < 16) {
    atomicAdd(&ws[0], a_mse);
    atomicAdd(&ws[1], a_msk);
    atomicAdd(&ws[2], a_dk);
    atomicAdd(&ws[3], a_dv);
    atomicAdd(&ws[4], a_k0);
    atomicAdd(&ws[5], a_v0);
  }
}

__global__ void pk_ws_zero(float* ws) {
  if (threadIdx.x < 8) ws[threadIdx.x] = 0.0f;
}

__global__ void pk_finalize(const float* __restrict__ ws, float* __restrict__ out) {
  const float Bf = (float)BSZ;
  const float mse  = ws[0] / ws[1];
  const float loss = mse
                   + (ws[4] / Bf) * SCALEF
                   + (ws[5] / Bf) * SCALEF
                   + (ws[2] / (Bf * (float)(TSZ - 1))) * SCALEF
                   + (ws[3] / (Bf * (float)(TSZ - 1))) * SCALEF;
  out[3 * BSZ * TSZ]     = loss;
  out[3 * BSZ * TSZ + 1] = mse;
}

extern "C" void kernel_launch(void* const* d_in, const int* in_sizes, int n_in,
                              void* d_out, int out_size, void* d_ws, size_t ws_size,
                              hipStream_t stream) {
  const float* cont  = (const float*)d_in[0];
  const int*   cat   = (const int*)  d_in[1];
  const float* lb    = (const float*)d_in[2];
  const float* dose  = (const float*)d_in[3];
  const float* td    = (const float*)d_in[4];
  const float* vfw   = (const float*)d_in[5];
  const float* vel   = (const float*)d_in[6];
  const float* embed = (const float*)d_in[7];
  const float* W_ih  = (const float*)d_in[8];
  const float* W_hh  = (const float*)d_in[9];
  const float* b_ih  = (const float*)d_in[10];
  const float* b_hh  = (const float*)d_in[11];
  const float* W_out = (const float*)d_in[12];
  const float* b_out = (const float*)d_in[13];

  float* out  = (float*)d_out;
  float* ws   = (float*)d_ws;
  float* outK = out;
  float* outV = out + (long)BSZ * TSZ;
  float* outC = out + 2L * BSZ * TSZ;

  pk_ws_zero<<<1, 32, 0, stream>>>(ws);
  pk_gru_scan<<<BSZ / 16, 32, 0, stream>>>(cont, cat, lb, dose, td, vfw, vel,
                                           embed, W_ih, W_hh, b_ih, b_hh,
                                           W_out, b_out, outK, outV, outC, ws);
  pk_finalize<<<1, 1, 0, stream>>>(ws, out);
}
